// TransformerExecutor_86380382257331
// MI455X (gfx1250) — compile-verified, matrix operand order checked by reading, not run on hardware
//
#include <hip/hip_runtime.h>

// ---------------------------------------------------------------------------
// Transformer decoder stack for MI455X (gfx1250, wave32, WMMA, async-LDS).
// One-time fp32->bf16 conversion (weights pre-transposed), then all GEMM /
// attention math in v_wmma_f32_16x16x32_bf16 with double-buffered
// GLOBAL_LOAD_ASYNC_TO_LDS staging (copies hidden behind WMMA).
// Softmax / LayerNorm in fp32. B=8 T=1024 D=512 H=8 hd=64 F=2048 NB=2.
// ---------------------------------------------------------------------------

typedef __attribute__((ext_vector_type(16))) __bf16 v16bf;
typedef __attribute__((ext_vector_type(8)))  float  v8f;

union BF16Frag { v16bf v; uint4 q[2]; };
union F32Acc   { v8f v; float f[8]; };
union Pack8    { uint4 q; unsigned short s[8]; };

static_assert(sizeof(BF16Frag) == 32, "frag size");
static_assert(sizeof(F32Acc)   == 32, "acc size");

__device__ __forceinline__ unsigned short f2bf(float f) {
  unsigned int u = __float_as_uint(f);
  u += 0x7FFFu + ((u >> 16) & 1u);   // round-to-nearest-even
  return (unsigned short)(u >> 16);
}

__device__ __forceinline__ v8f wmma_bf16(const v16bf& a, const v16bf& b, const v8f& c) {
  return __builtin_amdgcn_wmma_f32_16x16x32_bf16(false, a, false, b, (short)0, c,
                                                 false, false);
}

// LDS byte offset of a __shared__ object (flat LDS address low 32 bits).
__device__ __forceinline__ unsigned lds_off(const void* p) {
  return (unsigned)(size_t)p;
}

// Async copy 16B global -> LDS (ASYNCcnt tracked, GV mode).
__device__ __forceinline__ void async_b128(unsigned lds_byte, const void* gaddr) {
  asm volatile("global_load_async_to_lds_b128 %0, %1, off"
               :: "v"(lds_byte), "v"(gaddr)
               : "memory");
}

__device__ __forceinline__ void wait_async0() {
  asm volatile("s_wait_asynccnt 0x0" ::: "memory");
}

// ---------------------------------------------------------------------------
// fp32 -> bf16 (row-major), 8 elems/thread, packed 16B stores.
// ---------------------------------------------------------------------------
__global__ __launch_bounds__(256) void cvt_kernel(const float* __restrict__ in,
                                                  unsigned short* __restrict__ out) {
  const size_t i = ((size_t)blockIdx.x * 256 + threadIdx.x) * 8;
  const float4 a = *(const float4*)(in + i);
  const float4 b = *(const float4*)(in + i + 4);
  Pack8 p;
  p.s[0] = f2bf(a.x); p.s[1] = f2bf(a.y); p.s[2] = f2bf(a.z); p.s[3] = f2bf(a.w);
  p.s[4] = f2bf(b.x); p.s[5] = f2bf(b.y); p.s[6] = f2bf(b.z); p.s[7] = f2bf(b.w);
  *(uint4*)(out + i) = p.q;
}

// ---------------------------------------------------------------------------
// fp32 [K][N] -> bf16 transposed [N][K] (weights, one-time; K = 1<<kshift).
// ---------------------------------------------------------------------------
__global__ __launch_bounds__(256) void cvtT_kernel(const float* __restrict__ in,
                                                   unsigned short* __restrict__ out,
                                                   int kshift, int N) {
  const int idx = blockIdx.x * 256 + threadIdx.x;  // over N*K outputs
  const int n = idx >> kshift, k = idx & ((1 << kshift) - 1);
  out[idx] = f2bf(in[(size_t)k * N + n]);
}

// ---------------------------------------------------------------------------
// GEMM: C = op(scale * (A @ B) + bias).
// A: bf16 [M][K] row-major.  Bt: bf16 [N][K] (pre-transposed weights).
// Block 256 thr (8 waves), tile 128x128x32, wave tile 32x64.
// Double-buffered async staging: tile k+1 copies overlap tile k WMMAs.
// ---------------------------------------------------------------------------
template <bool HAS_BIAS, bool RELU, bool HAS_CF, bool HAS_CH>
__global__ __launch_bounds__(256) void gemm_bf16_kernel(
    const unsigned short* __restrict__ A, const unsigned short* __restrict__ Bt,
    const float* __restrict__ bias, float* __restrict__ Cf,
    unsigned short* __restrict__ Ch, float scale, int M, int N, int K) {
  __shared__ __align__(16) unsigned short As[2][128][40];  // 80B rows, 10240B/buf
  __shared__ __align__(16) unsigned short Bs[2][128][40];  // Bs[n][k]

  const int tid  = threadIdx.x;
  const int lane = tid & 31;
  const int wave = tid >> 5;
  const int wm   = wave >> 1;          // 0..3 (M)
  const int wn   = wave & 1;           // 0..1 (N)
  const int m0   = blockIdx.y * 128;
  const int n0   = blockIdx.x * 128;
  const int koff = (lane >> 4) * 8;
  const unsigned asB = lds_off(&As[0][0][0]);
  const unsigned bsB = lds_off(&Bs[0][0][0]);

  auto stage = [&](int buf, int k0) {
    const unsigned ao = asB + buf * 10240, bo = bsB + buf * 10240;
#pragma unroll
    for (int i = 0; i < 2; i++) {
      const int c = tid + i * 256;
      const int r = c >> 2, q = c & 3;
      async_b128(ao + r * 80 + q * 16, A  + (size_t)(m0 + r) * K + k0 + q * 8);
      async_b128(bo + r * 80 + q * 16, Bt + (size_t)(n0 + r) * K + k0 + q * 8);
    }
  };

  F32Acc acc[2][4];
#pragma unroll
  for (int s = 0; s < 2; s++)
#pragma unroll
    for (int t = 0; t < 4; t++)
#pragma unroll
      for (int r = 0; r < 8; r++) acc[s][t].f[r] = 0.0f;

  stage(0, 0);
  const int KT = K >> 5;
  for (int kt = 0; kt < KT; kt++) {
    const int cur = kt & 1;
    wait_async0();       // my copies into buf[cur] done
    __syncthreads();     // everyone's copies done; everyone's reads of buf[cur^1] done
    if (kt + 1 < KT) stage(cur ^ 1, (kt + 1) << 5);  // prefetch overlaps WMMA below

    BF16Frag af[2], bf[4];
#pragma unroll
    for (int s = 0; s < 2; s++) {
      const int mr = wm * 32 + s * 16 + (lane & 15);
      af[s].q[0] = *(const uint4*)&As[cur][mr][koff];
      af[s].q[1] = *(const uint4*)&As[cur][mr][16 + koff];
    }
#pragma unroll
    for (int t = 0; t < 4; t++) {
      const int nc = wn * 64 + t * 16 + (lane & 15);
      bf[t].q[0] = *(const uint4*)&Bs[cur][nc][koff];
      bf[t].q[1] = *(const uint4*)&Bs[cur][nc][16 + koff];
    }
#pragma unroll
    for (int s = 0; s < 2; s++)
#pragma unroll
      for (int t = 0; t < 4; t++)
        acc[s][t].v = wmma_bf16(af[s].v, bf[t].v, acc[s][t].v);
  }

  // epilogue: compile-time variant, hoisted bias, incremental row pointers
#pragma unroll
  for (int s = 0; s < 2; s++) {
    const int mb = m0 + wm * 32 + s * 16 + ((lane >> 4) << 3);
#pragma unroll
    for (int t = 0; t < 4; t++) {
      const int n = n0 + wn * 64 + t * 16 + (lane & 15);
      float bb = 0.0f;
      if (HAS_BIAS) bb = bias[n];
      float* pf = HAS_CF ? (Cf + (size_t)mb * N + n) : nullptr;
      unsigned short* ph = HAS_CH ? (Ch + (size_t)mb * N + n) : nullptr;
#pragma unroll
      for (int r = 0; r < 8; r++) {
        float v = acc[s][t].f[r] * scale + bb;
        if (RELU) v = fmaxf(v, 0.0f);
        if (HAS_CF) { *pf = v; pf += N; }
        if (HAS_CH) { *ph = f2bf(v); ph += N; }
      }
    }
  }
}

// ---------------------------------------------------------------------------
// Flash attention (bf16 in / bf16 out). hd=64, T=1024, D=512, H=8.
// Block = 128 thr (4 waves), grid = (T/64, H, B). Q pre-scaled by 1/sqrt(hd)
// in the projection GEMM. K/V tiles double-buffered: kb+1 staged (async K,
// transposed V) while kb's S / softmax / P@V runs.
// ---------------------------------------------------------------------------
__global__ __launch_bounds__(128) void attn_kernel(
    const unsigned short* __restrict__ Q, const unsigned short* __restrict__ Kp,
    const unsigned short* __restrict__ V, unsigned short* __restrict__ O,
    int causal) {
  __shared__ __align__(16) unsigned short Qs[64][72];      // [row][hd]
  __shared__ __align__(16) unsigned short Ks[2][64][72];   // [key][hd]
  __shared__ __align__(16) unsigned short Vts[2][64][72];  // [hd][key]
  __shared__ __align__(16) unsigned short Ps[64][72];      // [row][key] probs
  __shared__ float Sf[64][68];
  __shared__ float m_s[64], l_s[64], a_s[64];

  const int tid  = threadIdx.x;
  const int lane = tid & 31;
  const int wave = tid >> 5;
  const int qb = blockIdx.x, h = blockIdx.y, b = blockIdx.z;
  const size_t base = ((size_t)b * 1024) * 512 + (size_t)h * 64;
  const int mbase = wave * 16;
  const int koff  = (lane >> 4) * 8;
  const int hrow  = (lane >> 4) << 3;
  const unsigned qsB = lds_off(&Qs[0][0]);
  const unsigned ksB = lds_off(&Ks[0][0][0]);

  auto stageKV = [&](int buf, int kb) {
    const unsigned ko = ksB + buf * 9216;
#pragma unroll
    for (int i = 0; i < 4; i++) {
      const int c = tid + i * 128;
      const int r = c >> 3, q = c & 7;
      async_b128(ko + r * 144 + q * 16,
                 Kp + base + (size_t)(kb * 64 + r) * 512 + q * 8);
    }
#pragma unroll
    for (int i = 0; i < 8; i++) {
      const int c2 = tid + i * 128;
      const int r = c2 >> 4, cc = (c2 & 15) << 2;
      const uint2 u = *(const uint2*)(V + base + (size_t)(kb * 64 + r) * 512 + cc);
      Vts[buf][cc + 0][r] = (unsigned short)(u.x & 0xffff);
      Vts[buf][cc + 1][r] = (unsigned short)(u.x >> 16);
      Vts[buf][cc + 2][r] = (unsigned short)(u.y & 0xffff);
      Vts[buf][cc + 3][r] = (unsigned short)(u.y >> 16);
    }
  };

  // prologue: Q tile (async) + first K/V tile
#pragma unroll
  for (int i = 0; i < 4; i++) {
    const int c = tid + i * 128;
    const int r = c >> 3, q = c & 7;
    async_b128(qsB + r * 144 + q * 16,
               Q + base + (size_t)(qb * 64 + r) * 512 + q * 8);
  }
  stageKV(0, 0);
  if (tid < 64) { m_s[tid] = -1e30f; l_s[tid] = 0.0f; }
  F32Acc o[4];
#pragma unroll
  for (int t = 0; t < 4; t++)
#pragma unroll
    for (int r = 0; r < 8; r++) o[t].f[r] = 0.0f;
  wait_async0();
  __syncthreads();   // Qs, Ks[0], Vts[0] ready

  BF16Frag qf[2];
  {
    const int mr = mbase + (lane & 15);
#pragma unroll
    for (int c = 0; c < 2; c++) {
      qf[c].q[0] = *(const uint4*)&Qs[mr][c * 32 + koff];
      qf[c].q[1] = *(const uint4*)&Qs[mr][c * 32 + 16 + koff];
    }
  }

  const int nkb = causal ? (qb + 1) : 16;
  for (int kb = 0; kb < nkb; kb++) {
    const int cur = kb & 1;
    if (kb + 1 < nkb) stageKV(cur ^ 1, kb + 1);  // overlaps everything below

    // S = Q @ K^T (this wave's 16 rows x 64 keys)
#pragma unroll
    for (int t = 0; t < 4; t++) {
      BF16Frag kf0, kf1;
      const int nk = t * 16 + (lane & 15);
      kf0.q[0] = *(const uint4*)&Ks[cur][nk][koff];
      kf0.q[1] = *(const uint4*)&Ks[cur][nk][16 + koff];
      kf1.q[0] = *(const uint4*)&Ks[cur][nk][32 + koff];
      kf1.q[1] = *(const uint4*)&Ks[cur][nk][48 + koff];
      F32Acc s;
#pragma unroll
      for (int r = 0; r < 8; r++) s.f[r] = 0.0f;
      s.v = wmma_bf16(qf[0].v, kf0.v, s.v);
      s.v = wmma_bf16(qf[1].v, kf1.v, s.v);
#pragma unroll
      for (int r = 0; r < 8; r++)
        Sf[mbase + r + hrow][t * 16 + (lane & 15)] = s.f[r];
    }
    __syncthreads();

    // online softmax, one thread per query row
    if (tid < 64) {
      const int row = tid;
      const int gi = qb * 64 + row;
      int jmax = causal ? (gi - kb * 64 + 1) : 64;
      if (jmax > 64) jmax = 64;
      const float mp = m_s[row];
      float rmax = -1e30f;
      for (int j = 0; j < jmax; j++) rmax = fmaxf(rmax, Sf[row][j]);
      const float nm = fmaxf(mp, rmax);
      const float alpha = __expf(mp - nm);
      float sum = 0.0f;
      for (int j = 0; j < 64; j++) {
        const float p = (j < jmax) ? __expf(Sf[row][j] - nm) : 0.0f;
        Ps[row][j] = f2bf(p);
        sum += p;
      }
      l_s[row] = alpha * l_s[row] + sum;
      m_s[row] = nm;
      a_s[row] = alpha;
    }
    __syncthreads();

    // rescale O and accumulate P @ V
    float al[8];
#pragma unroll
    for (int r = 0; r < 8; r++) al[r] = a_s[mbase + r + hrow];
#pragma unroll
    for (int t = 0; t < 4; t++)
#pragma unroll
      for (int r = 0; r < 8; r++) o[t].f[r] *= al[r];

    BF16Frag pf[2];
    {
      const int mr = mbase + (lane & 15);
      pf[0].q[0] = *(const uint4*)&Ps[mr][koff];
      pf[0].q[1] = *(const uint4*)&Ps[mr][16 + koff];
      pf[1].q[0] = *(const uint4*)&Ps[mr][32 + koff];
      pf[1].q[1] = *(const uint4*)&Ps[mr][48 + koff];
    }
#pragma unroll
    for (int t = 0; t < 4; t++) {
      BF16Frag vf0, vf1;
      const int ne = t * 16 + (lane & 15);
      vf0.q[0] = *(const uint4*)&Vts[cur][ne][koff];
      vf0.q[1] = *(const uint4*)&Vts[cur][ne][16 + koff];
      vf1.q[0] = *(const uint4*)&Vts[cur][ne][32 + koff];
      vf1.q[1] = *(const uint4*)&Vts[cur][ne][48 + koff];
      o[t].v = wmma_bf16(pf[0].v, vf0.v, o[t].v);
      o[t].v = wmma_bf16(pf[1].v, vf1.v, o[t].v);
    }

    wait_async0();     // staged K copies for kb+1 complete
    __syncthreads();   // all reads of buf[cur] done; staged V stores visible
  }

  if (tid < 64) a_s[tid] = 1.0f / l_s[tid];
  __syncthreads();

  float inv[8];
#pragma unroll
  for (int r = 0; r < 8; r++) inv[r] = a_s[mbase + hrow + r];
#pragma unroll
  for (int t = 0; t < 4; t++) {
    const int e = t * 16 + (lane & 15);
    unsigned short* p = O + base + (size_t)(qb * 64 + mbase + hrow) * 512 + e;
#pragma unroll
    for (int r = 0; r < 8; r++) {
      *p = f2bf(o[t].f[r] * inv[r]);
      p += 512;
    }
  }
}

// ---------------------------------------------------------------------------
// LayerNorm with residual: LN(a + res) * g + beta -> fp32 out + bf16 out16.
// One wave32 per row (D = 512); 8 rows per 256-thread block.
// ---------------------------------------------------------------------------
__global__ __launch_bounds__(256) void ln_kernel(
    const float* __restrict__ a, const float* __restrict__ res,
    const float* __restrict__ g, const float* __restrict__ beta,
    float* __restrict__ out, unsigned short* __restrict__ out16) {
  const int lane = threadIdx.x & 31;
  const int wave = threadIdx.x >> 5;
  const int row  = blockIdx.x * 8 + wave;
  const size_t base = (size_t)row * 512;

  float x[16];
  float s = 0.0f;
#pragma unroll
  for (int j = 0; j < 16; j++) {
    const int e = lane + j * 32;
    x[j] = a[base + e] + res[base + e];
    s += x[j];
  }
#pragma unroll
  for (int off = 16; off > 0; off >>= 1) s += __shfl_xor(s, off, 32);
  const float mu = s * (1.0f / 512.0f);

  float v = 0.0f;
#pragma unroll
  for (int j = 0; j < 16; j++) {
    const float d = x[j] - mu;
    v += d * d;
  }
#pragma unroll
  for (int off = 16; off > 0; off >>= 1) v += __shfl_xor(v, off, 32);
  const float rs = rsqrtf(v * (1.0f / 512.0f) + 1e-5f);

#pragma unroll
  for (int j = 0; j < 16; j++) {
    const int e = lane + j * 32;
    const float y = (x[j] - mu) * rs * g[e] + beta[e];
    out[base + e] = y;
    out16[base + e] = f2bf(y);
  }
}

// ---------------------------------------------------------------------------
// Host-side orchestration
// ---------------------------------------------------------------------------
static inline void launch_gemm(hipStream_t s, const unsigned short* A,
                               const unsigned short* Bt, const float* bias,
                               float* Cf, unsigned short* Ch, float scale,
                               int M, int N, int K, int relu) {
  dim3 grid(N / 128, M / 128), blk(256);
  if (bias) {
    if (relu)
      gemm_bf16_kernel<true, true, false, true><<<grid, blk, 0, s>>>(A, Bt, bias, Cf, Ch, scale, M, N, K);
    else
      gemm_bf16_kernel<true, false, true, false><<<grid, blk, 0, s>>>(A, Bt, bias, Cf, Ch, scale, M, N, K);
  } else if (Cf && Ch) {
    gemm_bf16_kernel<false, false, true, true><<<grid, blk, 0, s>>>(A, Bt, bias, Cf, Ch, scale, M, N, K);
  } else if (Cf) {
    gemm_bf16_kernel<false, false, true, false><<<grid, blk, 0, s>>>(A, Bt, bias, Cf, Ch, scale, M, N, K);
  } else {
    gemm_bf16_kernel<false, false, false, true><<<grid, blk, 0, s>>>(A, Bt, bias, Cf, Ch, scale, M, N, K);
  }
}

extern "C" void kernel_launch(void* const* d_in, const int* in_sizes, int n_in,
                              void* d_out, int out_size, void* d_ws, size_t ws_size,
                              hipStream_t stream) {
  (void)in_sizes; (void)n_in; (void)out_size; (void)ws_size;

  const float* q_in = (const float*)d_in[0];
  const float* k_in = (const float*)d_in[1];
  const float* v_in = (const float*)d_in[2];
  const float* enc  = (const float*)d_in[3];
  const float* wq_s = (const float*)d_in[4];
  const float* wk_s = (const float*)d_in[5];
  const float* wv_s = (const float*)d_in[6];
  const float* wo_s = (const float*)d_in[7];
  const float* wq_c = (const float*)d_in[8];
  const float* wk_c = (const float*)d_in[9];
  const float* wv_c = (const float*)d_in[10];
  const float* wo_c = (const float*)d_in[11];
  const float* ln1g = (const float*)d_in[12];
  const float* ln1b = (const float*)d_in[13];
  const float* w1   = (const float*)d_in[14];
  const float* b1   = (const float*)d_in[15];
  const float* w2   = (const float*)d_in[16];
  const float* b2   = (const float*)d_in[17];
  const float* ln2g = (const float*)d_in[18];
  const float* ln2b = (const float*)d_in[19];
  // d_in[20] = n_heads (hardcoded 8 / hd 64)

  const int M = 8 * 1024, D = 512, F = 2048;
  const size_t NTD = (size_t)M * D;          // 4,194,304
  const size_t DD  = (size_t)D * D;          // 262,144
  const size_t DF  = (size_t)D * F;          // 1,048,576

  // ---- workspace carve (bf16 pool, then fp32 pool) ----
  unsigned short* u = (unsigned short*)d_ws;
  auto takeU = [&](size_t n) { unsigned short* p = u; u += n; return p; };
  unsigned short* wqst = takeU(2 * DD);
  unsigned short* wkst = takeU(2 * DD);
  unsigned short* wvst = takeU(2 * DD);
  unsigned short* wost = takeU(2 * DD);
  unsigned short* wqct = takeU(2 * DD);
  unsigned short* wkct = takeU(2 * DD);
  unsigned short* wvct = takeU(2 * DD);
  unsigned short* woct = takeU(2 * DD);
  unsigned short* w1t  = takeU(2 * DF);      // [F][D] per block
  unsigned short* w2t  = takeU(2 * DF);      // [D][F] per block
  unsigned short* q16  = takeU(NTD);
  unsigned short* k16  = takeU(NTD);
  unsigned short* v16  = takeU(NTD);
  unsigned short* e16  = takeU(NTD);
  unsigned short* x16  = takeU(NTD);         // next-block input
  unsigned short* pA16 = takeU(NTD);         // Qp / Qc
  unsigned short* pB16 = takeU(NTD);         // Kp / Kc
  unsigned short* pC16 = takeU(NTD);         // Vp / Vc
  unsigned short* at16 = takeU(NTD);         // attention out
  unsigned short* mo16 = takeU(NTD);
  unsigned short* x116 = takeU(NTD);
  unsigned short* h116 = takeU((size_t)M * F);
  float* fpool = (float*)u;
  float* mo_f = fpool;
  float* a_f  = fpool + NTD;
  float* x1_f = fpool + 2 * NTD;
  float* h2_f = fpool + 3 * NTD;

  // ---- one-time conversions ----
  const dim3 blk(256);
  cvt_kernel<<<dim3(NTD / 2048), blk, 0, stream>>>(q_in, q16);
  cvt_kernel<<<dim3(NTD / 2048), blk, 0, stream>>>(k_in, k16);
  cvt_kernel<<<dim3(NTD / 2048), blk, 0, stream>>>(v_in, v16);
  cvt_kernel<<<dim3(NTD / 2048), blk, 0, stream>>>(enc, e16);
  for (int i = 0; i < 2; i++) {
    const size_t oDD = (size_t)i * DD, oDF = (size_t)i * DF;
    cvtT_kernel<<<dim3(DD / 256), blk, 0, stream>>>(wq_s + oDD, wqst + oDD, 9, D);
    cvtT_kernel<<<dim3(DD / 256), blk, 0, stream>>>(wk_s + oDD, wkst + oDD, 9, D);
    cvtT_kernel<<<dim3(DD / 256), blk, 0, stream>>>(wv_s + oDD, wvst + oDD, 9, D);
    cvtT_kernel<<<dim3(DD / 256), blk, 0, stream>>>(wo_s + oDD, wost + oDD, 9, D);
    cvtT_kernel<<<dim3(DD / 256), blk, 0, stream>>>(wq_c + oDD, wqct + oDD, 9, D);
    cvtT_kernel<<<dim3(DD / 256), blk, 0, stream>>>(wk_c + oDD, wkct + oDD, 9, D);
    cvtT_kernel<<<dim3(DD / 256), blk, 0, stream>>>(wv_c + oDD, wvct + oDD, 9, D);
    cvtT_kernel<<<dim3(DD / 256), blk, 0, stream>>>(wo_c + oDD, woct + oDD, 9, D);
    cvtT_kernel<<<dim3(DF / 256), blk, 0, stream>>>(w1 + oDF, w1t + oDF, 9, F);  // [F][D], K=D
    cvtT_kernel<<<dim3(DF / 256), blk, 0, stream>>>(w2 + oDF, w2t + oDF, 11, D); // [D][F], K=F
  }

  const dim3 attn_grid(16, 8, 8), attn_blk(128);
  const dim3 ln_grid(1024), ln_blk(256);
  const float qscale = 0.125f;  // 1/sqrt(64)

  for (int i = 0; i < 2; i++) {
    const size_t oDD = (size_t)i * DD;
    const size_t oD  = (size_t)i * D;
    const size_t oDF = (size_t)i * DF;
    const size_t oF  = (size_t)i * F;

    const unsigned short* xq = (i == 0) ? q16 : x16;
    const unsigned short* xk = (i == 0) ? k16 : x16;
    const unsigned short* xv = (i == 0) ? v16 : x16;

    // --- self attention (causal); Q pre-scaled by 1/sqrt(hd) ---
    launch_gemm(stream, xq, wqst + oDD, nullptr, nullptr, pA16, qscale, M, D, D, 0);
    launch_gemm(stream, xk, wkst + oDD, nullptr, nullptr, pB16, 1.0f, M, D, D, 0);
    launch_gemm(stream, xv, wvst + oDD, nullptr, nullptr, pC16, 1.0f, M, D, D, 0);
    attn_kernel<<<attn_grid, attn_blk, 0, stream>>>(pA16, pB16, pC16, at16, 1);
    launch_gemm(stream, at16, wost + oDD, nullptr, mo_f, mo16, 1.0f, M, D, D, 0);

    // --- cross attention (enc is Q and K, mo is V; no mask) ---
    launch_gemm(stream, e16, wqct + oDD, nullptr, nullptr, pA16, qscale, M, D, D, 0);
    launch_gemm(stream, e16, wkct + oDD, nullptr, nullptr, pB16, 1.0f, M, D, D, 0);
    launch_gemm(stream, mo16, wvct + oDD, nullptr, nullptr, pC16, 1.0f, M, D, D, 0);
    attn_kernel<<<attn_grid, attn_blk, 0, stream>>>(pA16, pB16, pC16, at16, 0);
    launch_gemm(stream, at16, woct + oDD, nullptr, a_f, nullptr, 1.0f, M, D, D, 0);

    // --- LN1(a + mo) -> x1 (fp32 + bf16) ---
    ln_kernel<<<ln_grid, ln_blk, 0, stream>>>(a_f, mo_f, ln1g + oD, ln1b + oD,
                                              x1_f, x116);

    // --- FFN ---
    launch_gemm(stream, x116, w1t + oDF, b1 + oF, nullptr, h116, 1.0f, M, F, D, 1);
    launch_gemm(stream, h116, w2t + oDF, b2 + oD, h2_f, nullptr, 1.0f, M, D, F, 0);

    // --- LN2(h2 + x1) -> next x (bf16) / final output (fp32) ---
    float* xout = (i == 1) ? (float*)d_out : h2_f;  // fp32 side unused when i==0
    ln_kernel<<<ln_grid, ln_blk, 0, stream>>>(h2_f, x1_f, ln2g + oD, ln2b + oD,
                                              xout, x16);
  }
}
